// PathGenerator_44470091383438
// MI455X (gfx1250) — matrix operand order repaired and checked by reading, not compiled, not run on hardware
//
#include <hip/hip_runtime.h>
#include <math.h>

typedef __attribute__((ext_vector_type(16))) __bf16 v16bf;
typedef __attribute__((ext_vector_type(8)))  float  v8f;

union Frag16 { v16bf v; unsigned int u[8]; };

__device__ __forceinline__ unsigned short f2bf(float x){
    unsigned int u = __float_as_uint(x);
    u += 0x7FFFu + ((u >> 16) & 1u);
    return (unsigned short)(u >> 16);
}
__device__ __forceinline__ float bf2f(unsigned short h){
    return __uint_as_float(((unsigned int)h) << 16);
}
__device__ __forceinline__ float sigmoidf_(float x){ return 1.f / (1.f + expf(-x)); }
__device__ __forceinline__ float lrelu02(float x){ return x > 0.f ? x : 0.2f * x; }

// ---------------------------------------------------------------------------
// f32 -> bf16 convert with row padding (zero rows beyond `rows`)
// ---------------------------------------------------------------------------
__global__ void k_cvt(const float* __restrict__ src, unsigned short* __restrict__ dst,
                      int rows, int srcCols, int copyCols, int dstLd, int padRowsTo)
{
    long long tot = (long long)padRowsTo * copyCols;
    long long idx = (long long)blockIdx.x * blockDim.x + threadIdx.x;
    if (idx >= tot) return;
    int r = (int)(idx / copyCols);
    int c = (int)(idx % copyCols);
    float v = (r < rows) ? src[(size_t)r * srcCols + c] : 0.f;
    dst[(size_t)r * dstLd + c] = f2bf(v);
}

__global__ void k_zerof(float* p, int n){
    int i = blockIdx.x * blockDim.x + threadIdx.x;
    if (i < n) p[i] = 0.f;
}
__global__ void k_zerob16(unsigned short* p, long long n){
    long long i = (long long)blockIdx.x * blockDim.x + threadIdx.x;
    if (i < n) p[i] = 0;
}
__global__ void k_addvec(const float* a, const float* b, float* o, int n){
    int i = blockIdx.x * blockDim.x + threadIdx.x;
    if (i < n) o[i] = a[i] + b[i];
}

// ---------------------------------------------------------------------------
// bf16 WMMA GEMM with async-DMA LDS staging (double buffered):
//   C(M,N) = A(M,K) * B(N,K)^T      (B row-major N x K)
// Block = 256 threads (8 waves), block tile 64x128, wave tile 32x32,
// K-step 32. A tile 64x32 + B tile 128x32 staged to LDS with
// global_load_async_to_lds_b128 (ASYNCcnt), overlapped with WMMA compute.
// A rows padded to gridDim.y*64 (zero-filled), K multiple of 32,
// B rows cover gridDim.x*128 columns. All tiles 16B aligned.
// ---------------------------------------------------------------------------
__global__ __launch_bounds__(256) void k_gemm_bf16(
    const unsigned short* __restrict__ A, int lda,
    const unsigned short* __restrict__ B, int ldb,
    float* __restrict__ Cf, unsigned short* __restrict__ Cb, int ldc,
    const float* __restrict__ bias, int M, int N, int K, int doRelu)
{
    __shared__ unsigned short sA[2][64][32];    //  8 KB
    __shared__ unsigned short sB[2][128][32];   // 16 KB

    const int t    = threadIdx.x;
    const int lane = t & 31;
    const int wv   = t >> 5;
    const int hl   = lane >> 4;     // half-wave
    const int l15  = lane & 15;
    const int rowBase = blockIdx.y * 64;
    const int colBase = blockIdx.x * 128;
    const int rm = (wv >> 2) * 32;  // wave tile origin within block
    const int nb = (wv & 3) * 32;

    // staging work split: A = 256 x 16B units (1/thread), B = 512 units (2/thread)
    const int ar = t >> 2, as = (t & 3) * 8;

    auto stage = [&](int kb, int buf) {
        {
            const unsigned short* g = A + (size_t)(rowBase + ar) * lda + kb + as;
            unsigned lo = (unsigned)(unsigned long long)&sA[buf][ar][as];
            asm volatile("global_load_async_to_lds_b128 %0, %1, off"
                         :: "v"(lo), "v"(g) : "memory");
        }
        #pragma unroll
        for (int i = 0; i < 2; ++i) {
            int idx = t + i * 256;
            int r = idx >> 2, s2 = (idx & 3) * 8;
            const unsigned short* g = B + (size_t)(colBase + r) * ldb + kb + s2;
            unsigned lo = (unsigned)(unsigned long long)&sB[buf][r][s2];
            asm volatile("global_load_async_to_lds_b128 %0, %1, off"
                         :: "v"(lo), "v"(g) : "memory");
        }
    };

    v8f acc00 = {}; v8f acc01 = {}; v8f acc10 = {}; v8f acc11 = {};

    stage(0, 0);
    asm volatile("s_wait_asynccnt 0x0" ::: "memory");
    __syncthreads();

    const int nk = K >> 5;
    for (int i = 0; i < nk; ++i) {
        const int buf = i & 1;
        if (i + 1 < nk) stage((i + 1) << 5, buf ^ 1);

        Frag16 a0, a1, b0, b1;
        #pragma unroll
        for (int q = 0; q < 4; ++q) {
            a0.u[q]     = *(const unsigned int*)&sA[buf][rm + l15     ][hl * 8 + 2 * q];
            a0.u[4 + q] = *(const unsigned int*)&sA[buf][rm + l15     ][16 + hl * 8 + 2 * q];
            a1.u[q]     = *(const unsigned int*)&sA[buf][rm + 16 + l15][hl * 8 + 2 * q];
            a1.u[4 + q] = *(const unsigned int*)&sA[buf][rm + 16 + l15][16 + hl * 8 + 2 * q];
        }
        #pragma unroll
        for (int q = 0; q < 8; ++q) {
            b0.u[q] = *(const unsigned int*)&sB[buf][nb + l15     ][(q >> 2) * 16 + hl * 8 + 2 * (q & 3)];
            b1.u[q] = *(const unsigned int*)&sB[buf][nb + 16 + l15][(q >> 2) * 16 + hl * 8 + 2 * (q & 3)];
        }
        acc00 = __builtin_amdgcn_wmma_f32_16x16x32_bf16(false, a0.v, false, b0.v, (short)0, acc00, false, false);
        acc01 = __builtin_amdgcn_wmma_f32_16x16x32_bf16(false, a0.v, false, b1.v, (short)0, acc01, false, false);
        acc10 = __builtin_amdgcn_wmma_f32_16x16x32_bf16(false, a1.v, false, b0.v, (short)0, acc10, false, false);
        acc11 = __builtin_amdgcn_wmma_f32_16x16x32_bf16(false, a1.v, false, b1.v, (short)0, acc11, false, false);

        if (i + 1 < nk) asm volatile("s_wait_asynccnt 0x0" ::: "memory");
        __syncthreads();
    }

    auto storeTile = [&](v8f acc, int r0, int c0) {
        int col = colBase + c0 + l15;
        if (col >= N) return;
        float bv = bias ? bias[col] : 0.f;
        #pragma unroll
        for (int r = 0; r < 8; ++r) {
            int row = rowBase + r0 + hl * 8 + r;
            if (row < M) {
                float v = acc[r] + bv;
                if (doRelu) v = fmaxf(v, 0.f);
                size_t o = (size_t)row * ldc + col;
                if (Cf) Cf[o] = v;
                if (Cb) Cb[o] = f2bf(v);
            }
        }
    };
    storeTile(acc00, rm,      nb);
    storeTile(acc01, rm,      nb + 16);
    storeTile(acc10, rm + 16, nb);
    storeTile(acc11, rm + 16, nb + 16);
}

// ---------------------------------------------------------------------------
// GAT attention scores: s[j,h] = h[j,h,:].a_src[h], d[j,h] = h[j,h,:].a_dst[h]
// One wave per row.
// ---------------------------------------------------------------------------
__global__ void k_gat_sd(const void* __restrict__ hmat, int isBf16, int ld, int nrows,
                         int heads, int hc,
                         const float* __restrict__ asrc, const float* __restrict__ adst,
                         float* __restrict__ s, float* __restrict__ d)
{
    int lane = threadIdx.x & 31;
    int row  = (blockIdx.x * blockDim.x + threadIdx.x) >> 5;
    if (row >= nrows) return;
    const unsigned short* rb = (const unsigned short*)hmat + (size_t)row * ld;
    const float*          rf = (const float*)hmat          + (size_t)row * ld;
    for (int h = 0; h < heads; ++h) {
        float ps = 0.f, pd = 0.f;
        int base = h * hc;
        for (int c = lane; c < hc; c += 32) {
            float v = isBf16 ? bf2f(rb[base + c]) : rf[base + c];
            ps += v * asrc[base + c];
            pd += v * adst[base + c];
        }
        for (int off = 16; off > 0; off >>= 1) {
            ps += __shfl_down(ps, off, 32);
            pd += __shfl_down(pd, off, 32);
        }
        if (lane == 0) {
            s[(size_t)row * heads + h] = ps;
            d[(size_t)row * heads + h] = pd;
        }
    }
}

// node-0 softmax stats (max & denom over all n incoming edges), per head
__global__ __launch_bounds__(1024) void k_gat_stats(const float* __restrict__ s,
                                                    const float* __restrict__ d,
                                                    int n, int heads, float* __restrict__ stats)
{
    __shared__ float red[1024];
    int tid = threadIdx.x;
    for (int h = 0; h < heads; ++h) {
        float d0 = d[h];
        float lm = -1e30f;
        for (int j = tid; j < n; j += 1024) {
            float e = lrelu02(s[(size_t)j * heads + h] + d0);
            lm = fmaxf(lm, e);
        }
        red[tid] = lm; __syncthreads();
        for (int off = 512; off > 0; off >>= 1) { if (tid < off) red[tid] = fmaxf(red[tid], red[tid + off]); __syncthreads(); }
        float m = red[0]; __syncthreads();
        float ls = 0.f;
        for (int j = tid; j < n; j += 1024) {
            float e = lrelu02(s[(size_t)j * heads + h] + d0);
            ls += expf(e - m);
        }
        red[tid] = ls; __syncthreads();
        for (int off = 512; off > 0; off >>= 1) { if (tid < off) red[tid] += red[tid + off]; __syncthreads(); }
        if (tid == 0) { stats[h * 2] = m; stats[h * 2 + 1] = red[0]; }
        __syncthreads();
    }
}

// GAT1 per-neighbor output (j >= 1): softmax over {center, self}, 4 heads
__global__ __launch_bounds__(256) void k_gat1_out(const unsigned short* __restrict__ h1,
                                                  const float* __restrict__ s, const float* __restrict__ d,
                                                  const float* __restrict__ b,
                                                  unsigned short* __restrict__ x1, int n)
{
    int j = blockIdx.x + 1;
    if (j >= n) return;
    __shared__ float a1s[4], a2s[4];
    if (threadIdx.x < 4) {
        int h = threadIdx.x;
        float dj = d[(size_t)j * 4 + h];
        float e1 = lrelu02(s[h] + dj);
        float e2 = lrelu02(s[(size_t)j * 4 + h] + dj);
        float m  = fmaxf(e1, e2);
        float w1 = expf(e1 - m), w2 = expf(e2 - m);
        float inv = 1.f / (w1 + w2);
        a1s[h] = w1 * inv; a2s[h] = w2 * inv;
    }
    __syncthreads();
    const unsigned short* h0 = h1;
    const unsigned short* hj = h1 + (size_t)j * 2048;
    unsigned short* xr = x1 + (size_t)j * 2048;
    for (int c = threadIdx.x; c < 2048; c += blockDim.x) {
        int h = c >> 9;
        float v = a1s[h] * bf2f(h0[c]) + a2s[h] * bf2f(hj[c]) + b[c];
        xr[c] = f2bf(fmaxf(v, 0.f));
    }
}

// GAT1 node-0 aggregation: acc[c] += sum_j alpha0_j * h1[j,c]  (includes j=0 self)
__global__ __launch_bounds__(256) void k_gat1_acc(const unsigned short* __restrict__ h1,
                                                  const float* __restrict__ s, const float* __restrict__ d,
                                                  const float* __restrict__ stats,
                                                  float* __restrict__ acc, int n)
{
    __shared__ float la[2048];
    for (int c = threadIdx.x; c < 2048; c += blockDim.x) la[c] = 0.f;
    __syncthreads();
    for (int j = blockIdx.x; j < n; j += gridDim.x) {
        float al[4];
        #pragma unroll
        for (int h = 0; h < 4; ++h) {
            float e = lrelu02(s[(size_t)j * 4 + h] + d[h]);
            al[h] = expf(e - stats[h * 2]) / stats[h * 2 + 1];
        }
        const unsigned short* hj = h1 + (size_t)j * 2048;
        for (int c = threadIdx.x; c < 2048; c += blockDim.x)
            la[c] += al[c >> 9] * bf2f(hj[c]);
    }
    __syncthreads();
    for (int c = threadIdx.x; c < 2048; c += blockDim.x) atomicAdd(&acc[c], la[c]);
}

__global__ void k_gat1_node0(const float* __restrict__ acc, const float* __restrict__ b,
                             unsigned short* __restrict__ x1)
{
    for (int c = threadIdx.x; c < 2048; c += blockDim.x)
        x1[c] = f2bf(fmaxf(acc[c] + b[c], 0.f));
}

// GAT2 node-0 aggregation (single head, f32 h2, 512 channels)
__global__ __launch_bounds__(256) void k_gat2_acc(const float* __restrict__ h2,
                                                  const float* __restrict__ s, const float* __restrict__ d,
                                                  const float* __restrict__ stats,
                                                  float* __restrict__ acc, int n)
{
    __shared__ float la[512];
    for (int c = threadIdx.x; c < 512; c += blockDim.x) la[c] = 0.f;
    __syncthreads();
    for (int j = blockIdx.x; j < n; j += gridDim.x) {
        float e  = lrelu02(s[j] + d[0]);
        float al = expf(e - stats[0]) / stats[1];
        const float* hj = h2 + (size_t)j * 512;
        for (int c = threadIdx.x; c < 512; c += blockDim.x)
            la[c] += al * hj[c];
    }
    __syncthreads();
    for (int c = threadIdx.x; c < 512; c += blockDim.x) atomicAdd(&acc[c], la[c]);
}

__global__ void k_gat2_fin(const float* __restrict__ acc, const float* __restrict__ b,
                           float* __restrict__ erep)
{
    int t = threadIdx.x;
    if (t < 512) erep[t] = fmaxf(acc[t] + b[t], 0.f);
}

// ---------------------------------------------------------------------------
// LSTM recurrence: one persistent workgroup (32 waves). xproj holds
// lstm_in@W_ih^T + b_ih + b_hh. Thread t<512 computes gates i,g of hidden t;
// thread t>=512 computes f,o of hidden t-512. h lives in LDS.
// ---------------------------------------------------------------------------
__global__ __launch_bounds__(1024) void k_lstm(const float* __restrict__ xproj,
                                               const unsigned short* __restrict__ whh,
                                               float* __restrict__ hout)
{
    __shared__ float sh_h[512];
    __shared__ float sh_fo[1024];
    int t = threadIdx.x;
    int j = t & 511;
    const unsigned short* w0 = whh + (size_t)t * 512;
    const unsigned short* w1 = whh + (size_t)(t + 1024) * 512;
    float c = 0.f;
    if (t < 512) sh_h[t] = 0.f;
    __syncthreads();
    for (int step = 0; step < 1024; ++step) {
        const float* xr = xproj + (size_t)step * 2048;
        float z0 = xr[t];
        float z1 = xr[t + 1024];
        for (int k = 0; k < 512; k += 2) {
            unsigned int p0 = *(const unsigned int*)(w0 + k);
            unsigned int p1 = *(const unsigned int*)(w1 + k);
            float ha = sh_h[k], hb = sh_h[k + 1];
            z0 += bf2f((unsigned short)p0) * ha + bf2f((unsigned short)(p0 >> 16)) * hb;
            z1 += bf2f((unsigned short)p1) * ha + bf2f((unsigned short)(p1 >> 16)) * hb;
        }
        if (t >= 512) { sh_fo[j] = z0; sh_fo[512 + j] = z1; }
        __syncthreads();
        if (t < 512) {
            float zf = sh_fo[j], zo = sh_fo[512 + j];
            c = sigmoidf_(zf) * c + sigmoidf_(z0) * tanhf(z1);
            sh_h[j] = sigmoidf_(zo) * tanhf(c);
        }
        __syncthreads();
    }
    if (t < 512) hout[t] = sh_h[t];
}

// ---------------------------------------------------------------------------
// Small dense heads
// ---------------------------------------------------------------------------
__global__ __launch_bounds__(512) void k_qenc(const float* __restrict__ q,
    const float* __restrict__ w1, const float* __restrict__ b1,
    const float* __restrict__ w2, const float* __restrict__ b2, float* __restrict__ out)
{
    __shared__ float sq[768];
    __shared__ float sh1[512];
    int t = threadIdx.x;
    for (int i = t; i < 768; i += 512) sq[i] = q[i];
    __syncthreads();
    float a = b1[t];
    const float* wr = w1 + (size_t)t * 768;
    for (int k = 0; k < 768; ++k) a += wr[k] * sq[k];
    sh1[t] = fmaxf(a, 0.f);
    __syncthreads();
    float a2 = b2[t];
    const float* wr2 = w2 + (size_t)t * 512;
    for (int k = 0; k < 512; ++k) a2 += wr2[k] * sh1[k];
    out[t] = a2;
}

__global__ __launch_bounds__(512) void k_comb(const float* __restrict__ qv, const float* __restrict__ pv,
    const float* __restrict__ ev,
    const float* __restrict__ qw, const float* __restrict__ qb,
    const float* __restrict__ pw, const float* __restrict__ pb,
    const float* __restrict__ ew, const float* __restrict__ eb, float* __restrict__ comb)
{
    __shared__ float s[512];
    int t = threadIdx.x;
    s[t] = qv[t]; __syncthreads();
    float a = qb[t]; const float* w = qw + (size_t)t * 512;
    for (int k = 0; k < 512; ++k) a += w[k] * s[k];
    comb[t] = a; __syncthreads();
    s[t] = pv[t]; __syncthreads();
    a = pb[t]; w = pw + (size_t)t * 512;
    for (int k = 0; k < 512; ++k) a += w[k] * s[k];
    comb[512 + t] = a; __syncthreads();
    s[t] = ev[t]; __syncthreads();
    a = eb[t]; w = ew + (size_t)t * 512;
    for (int k = 0; k < 512; ++k) a += w[k] * s[k];
    comb[1024 + t] = a;
}

__global__ __launch_bounds__(512) void k_pol12(const float* __restrict__ comb,
    const float* __restrict__ w1, const float* __restrict__ b1,
    const float* __restrict__ w2, const float* __restrict__ b2, float* __restrict__ hdd2)
{
    __shared__ float sc[1536];
    __shared__ float sh[512];
    int t = threadIdx.x;
    for (int i = t; i < 1536; i += 512) sc[i] = comb[i];
    __syncthreads();
    float a = b1[t];
    const float* w = w1 + (size_t)t * 1536;
    for (int k = 0; k < 1536; ++k) a += w[k] * sc[k];
    sh[t] = fmaxf(a, 0.f);
    __syncthreads();
    if (t < 256) {
        float a2 = b2[t];
        const float* w2r = w2 + (size_t)t * 512;
        for (int k = 0; k < 512; ++k) a2 += w2r[k] * sh[k];
        hdd2[t] = fmaxf(a2, 0.f);
    }
}

// gathered policy head + softmax; out = [action_probs(512); valid_logits(512)]
__global__ __launch_bounds__(512) void k_pol3(const float* __restrict__ hdd2,
    const float* __restrict__ w3, const float* __restrict__ b3,
    const int* __restrict__ valid, float* __restrict__ out)
{
    __shared__ float sh[256];
    __shared__ float red[512];
    int t = threadIdx.x;
    if (t < 256) sh[t] = hdd2[t];
    __syncthreads();
    int r = valid[t];
    float a = b3[r];
    const float* w = w3 + (size_t)r * 256;
    for (int k = 0; k < 256; ++k) a += w[k] * sh[k];
    out[512 + t] = a;
    red[t] = a; __syncthreads();
    for (int off = 256; off > 0; off >>= 1) { if (t < off) red[t] = fmaxf(red[t], red[t + off]); __syncthreads(); }
    float m = red[0]; __syncthreads();
    float e = expf(a - m);
    red[t] = e; __syncthreads();
    for (int off = 256; off > 0; off >>= 1) { if (t < off) red[t] += red[t + off]; __syncthreads(); }
    out[t] = e / red[0];
}

// ---------------------------------------------------------------------------
extern "C" void kernel_launch(void* const* d_in, const int* in_sizes, int n_in,
                              void* d_out, int out_size, void* d_ws, size_t ws_size,
                              hipStream_t stream)
{
    const float* question = (const float*)d_in[0];
    const float* cur_ent  = (const float*)d_in[1];
    const float* path_ent = (const float*)d_in[2];
    const float* path_rel = (const float*)d_in[3];
    const float* nbr      = (const float*)d_in[4];
    const float* qw1 = (const float*)d_in[5];  const float* qb1 = (const float*)d_in[6];
    const float* qw2 = (const float*)d_in[7];  const float* qb2 = (const float*)d_in[8];
    const float* epw = (const float*)d_in[9];  const float* epb = (const float*)d_in[10];
    const float* wih = (const float*)d_in[11]; const float* whh = (const float*)d_in[12];
    const float* bih = (const float*)d_in[13]; const float* bhh = (const float*)d_in[14];
    const float* g1w = (const float*)d_in[15]; const float* g1as = (const float*)d_in[16];
    const float* g1ad = (const float*)d_in[17]; const float* g1b = (const float*)d_in[18];
    const float* g2w = (const float*)d_in[19]; const float* g2as = (const float*)d_in[20];
    const float* g2ad = (const float*)d_in[21]; const float* g2b = (const float*)d_in[22];
    const float* qpw = (const float*)d_in[23]; const float* qpb = (const float*)d_in[24];
    const float* ppw = (const float*)d_in[25]; const float* ppb = (const float*)d_in[26];
    const float* eepw = (const float*)d_in[27]; const float* eepb = (const float*)d_in[28];
    const float* pw1 = (const float*)d_in[29]; const float* pb1 = (const float*)d_in[30];
    const float* pw2 = (const float*)d_in[31]; const float* pb2 = (const float*)d_in[32];
    const float* pw3 = (const float*)d_in[33]; const float* pb3 = (const float*)d_in[34];
    const int*   valid = (const int*)d_in[35];
    float* out = (float*)d_out;

    const int NN = 16385;          // nodes (center + 16384 neighbors)
    const int NNp = 16448;         // padded to multiple of 64

    char* ws = (char*)d_ws;
    size_t off = 0;
    auto alloc = [&](size_t bytes) -> char* {
        char* p = ws + off;
        off += (bytes + 255) & ~(size_t)255;
        return p;
    };
    unsigned short* nodes_bf = (unsigned short*)alloc((size_t)NNp * 768 * 2);
    unsigned short* w1_bf    = (unsigned short*)alloc((size_t)2048 * 768 * 2);
    unsigned short* h1_bf    = (unsigned short*)alloc((size_t)NNp * 2048 * 2);
    unsigned short* x1_bf    = (unsigned short*)alloc((size_t)NNp * 2048 * 2);
    unsigned short* w2_bf    = (unsigned short*)alloc((size_t)512 * 2048 * 2);
    float*          h2       = (float*)alloc((size_t)NNp * 512 * 4);
    unsigned short* pe_bf    = (unsigned short*)alloc((size_t)1024 * 768 * 2);
    unsigned short* epw_bf   = (unsigned short*)alloc((size_t)768 * 768 * 2);
    unsigned short* lstm_in  = (unsigned short*)alloc((size_t)1024 * 1536 * 2);
    unsigned short* wih_bf   = (unsigned short*)alloc((size_t)2048 * 1536 * 2);
    unsigned short* whh_bf   = (unsigned short*)alloc((size_t)2048 * 512 * 2);
    float*          xproj    = (float*)alloc((size_t)1024 * 2048 * 4);
    float*          biassum  = (float*)alloc(2048 * 4);
    float*          s1       = (float*)alloc((size_t)NN * 4 * 4);
    float*          d1       = (float*)alloc((size_t)NN * 4 * 4);
    float*          stats1   = (float*)alloc(8 * 4);
    float*          acc0     = (float*)alloc(2048 * 4);
    float*          s2       = (float*)alloc((size_t)NN * 4);
    float*          d2       = (float*)alloc((size_t)NN * 4);
    float*          stats2   = (float*)alloc(2 * 4);
    float*          acc2     = (float*)alloc(512 * 4);
    float*          qvec     = (float*)alloc(512 * 4);
    float*          prep     = (float*)alloc(512 * 4);
    float*          erep     = (float*)alloc(512 * 4);
    float*          comb     = (float*)alloc(1536 * 4);
    float*          hdd2     = (float*)alloc(256 * 4);
    (void)ws_size; (void)in_sizes; (void)n_in; (void)out_size;

    auto cdiv = [](long long a, long long b) -> int { return (int)((a + b - 1) / b); };
    auto cvt = [&](const float* src, unsigned short* dst, int rows, int srcCols,
                   int copyCols, int dstLd, int padRowsTo) {
        long long tot = (long long)padRowsTo * copyCols;
        k_cvt<<<cdiv(tot, 256), 256, 0, stream>>>(src, dst, rows, srcCols, copyCols, dstLd, padRowsTo);
    };

    // ---- conversions / packing ----
    cvt(cur_ent, nodes_bf, 1, 768, 768, 768, 1);                       // node 0
    cvt(nbr, nodes_bf + 768, 16384, 768, 768, 768, NNp - 1);           // nodes 1.. + zero pad
    cvt(g1w, w1_bf, 2048, 768, 768, 768, 2048);
    cvt(g2w, w2_bf, 512, 2048, 2048, 2048, 512);
    cvt(path_ent, pe_bf, 1024, 768, 768, 768, 1024);                   // path_entities[:-1]
    cvt(epw, epw_bf, 768, 768, 768, 768, 768);
    cvt(path_rel, lstm_in + 768, 1024, 768, 768, 1536, 1024);          // right half of lstm_in
    cvt(wih, wih_bf, 2048, 1536, 1536, 1536, 2048);
    cvt(whh, whh_bf, 2048, 512, 512, 512, 2048);
    k_addvec<<<8, 256, 0, stream>>>(bih, bhh, biassum, 2048);

    // ---- question encoder ----
    k_qenc<<<1, 512, 0, stream>>>(question, qw1, qb1, qw2, qb2, qvec);

    // ---- ent_proj GEMM -> left half of lstm_in (bf16) ----
    k_gemm_bf16<<<dim3(6, 16), 256, 0, stream>>>(pe_bf, 768, epw_bf, 768,
        nullptr, lstm_in, 1536, epb, 1024, 768, 768, 0);

    // ---- LSTM input projection GEMM (biases folded) ----
    k_gemm_bf16<<<dim3(16, 16), 256, 0, stream>>>(lstm_in, 1536, wih_bf, 1536,
        xproj, nullptr, 2048, biassum, 1024, 2048, 1536, 0);

    // ---- LSTM recurrence ----
    k_lstm<<<1, 1024, 0, stream>>>(xproj, whh_bf, prep);

    // ---- GAT layer 1: h1 = nodes @ W1^T (bf16 out) ----
    k_gemm_bf16<<<dim3(16, 257), 256, 0, stream>>>(nodes_bf, 768, w1_bf, 768,
        nullptr, h1_bf, 2048, nullptr, NNp, 2048, 768, 0);

    k_gat_sd<<<cdiv((long long)NN * 32, 256), 256, 0, stream>>>(h1_bf, 1, 2048, NN, 4, 512, g1as, g1ad, s1, d1);
    k_gat_stats<<<1, 1024, 0, stream>>>(s1, d1, NN, 4, stats1);
    k_zerof<<<8, 256, 0, stream>>>(acc0, 2048);
    k_gat1_out<<<NN - 1, 256, 0, stream>>>(h1_bf, s1, d1, g1b, x1_bf, NN);
    k_gat1_acc<<<64, 256, 0, stream>>>(h1_bf, s1, d1, stats1, acc0, NN);
    k_gat1_node0<<<1, 256, 0, stream>>>(acc0, g1b, x1_bf);
    {
        long long padElems = (long long)(NNp - NN) * 2048;
        k_zerob16<<<cdiv(padElems, 256), 256, 0, stream>>>(x1_bf + (size_t)NN * 2048, padElems);
    }

    // ---- GAT layer 2: h2 = x1 @ W2^T (f32 out) ----
    k_gemm_bf16<<<dim3(4, 257), 256, 0, stream>>>(x1_bf, 2048, w2_bf, 2048,
        h2, nullptr, 512, nullptr, NN, 512, 2048, 0);

    k_gat_sd<<<cdiv((long long)NN * 32, 256), 256, 0, stream>>>(h2, 0, 512, NN, 1, 512, g2as, g2ad, s2, d2);
    k_gat_stats<<<1, 1024, 0, stream>>>(s2, d2, NN, 1, stats2);
    k_zerof<<<2, 256, 0, stream>>>(acc2, 512);
    k_gat2_acc<<<64, 256, 0, stream>>>(h2, s2, d2, stats2, acc2, NN);
    k_gat2_fin<<<1, 512, 0, stream>>>(acc2, g2b, erep);

    // ---- heads ----
    k_comb<<<1, 512, 0, stream>>>(qvec, prep, erep, qpw, qpb, ppw, ppb, eepw, eepb, comb);
    k_pol12<<<1, 512, 0, stream>>>(comb, pw1, pb1, pw2, pb2, hdd2);
    k_pol3<<<1, 512, 0, stream>>>(hdd2, pw3, pb3, valid, out);
}
